// SelfAttention_22539988369889
// MI455X (gfx1250) — matrix operand order repaired and checked by reading, not compiled, not run on hardware
//
#include <hip/hip_runtime.h>
#include <hip/hip_bf16.h>
#include <cstdint>
#include <cstddef>

// ---------------------------------------------------------------------------
// MI455X / gfx1250 fused self-attention:
//   x -> (Q,K,V) bf16 WMMA GEMMs -> flash attention (WMMA QK^T + PV) -> Wo GEMM
// Matrix math: v_wmma_f32_16x16x32_bf16 (wave32).
// Direct-copy tile staging: global_load_async_to_lds_b128 (ASYNCcnt).
// ---------------------------------------------------------------------------

typedef __attribute__((ext_vector_type(16))) __bf16 v16bf;
typedef __attribute__((ext_vector_type(8)))  __bf16 v8bf;
typedef __attribute__((ext_vector_type(8)))  float  v8f;

union V16 { v16bf v; v8bf h[2]; };

constexpr int   BATCH  = 4;
constexpr int   NSEQ   = 2048;
constexpr int   DMODEL = 1024;
constexpr int   NHEADS = 8;
constexpr int   HDIM   = 128;
constexpr int   PROJ   = 1024;
constexpr float SM_SCALE = 0.08838834764831845f;  // 1/sqrt(128)
constexpr float L2E      = 1.4426950408889634f;

static __device__ __forceinline__ v8f wmma_bf16(v16bf a, v16bf b, v8f c) {
  // D = A(16x32 bf16) * B(32x16 bf16) + C(16x16 f32)
  return __builtin_amdgcn_wmma_f32_16x16x32_bf16(
      /*neg_a=*/false, a, /*neg_b=*/false, b,
      /*c_mod=*/(short)0, c, /*reuse_a=*/false, /*reuse_b=*/false);
}

// Async direct copy: 64 contiguous bytes global -> LDS (4 x b128 per lane).
// lds_off = LDS byte address (addrspace-3 offset), g = global source.
static __device__ __forceinline__ void async_copy64(unsigned lds_off,
                                                    const void* g) {
  asm volatile(
      "global_load_async_to_lds_b128 %0, %1, off\n\t"
      "global_load_async_to_lds_b128 %0, %1, off offset:16\n\t"
      "global_load_async_to_lds_b128 %0, %1, off offset:32\n\t"
      "global_load_async_to_lds_b128 %0, %1, off offset:48"
      :: "v"(lds_off), "v"(g) : "memory");
}
static __device__ __forceinline__ void async_wait0() {
  asm volatile("s_wait_asynccnt 0x0" ::: "memory");
}

static __device__ __forceinline__ float half_max16(float v) {
  v = fmaxf(v, __shfl_xor(v, 1, 32));
  v = fmaxf(v, __shfl_xor(v, 2, 32));
  v = fmaxf(v, __shfl_xor(v, 4, 32));
  v = fmaxf(v, __shfl_xor(v, 8, 32));
  return v;
}
static __device__ __forceinline__ float half_sum16(float v) {
  v += __shfl_xor(v, 1, 32);
  v += __shfl_xor(v, 2, 32);
  v += __shfl_xor(v, 4, 32);
  v += __shfl_xor(v, 8, 32);
  return v;
}

// ---------------------------------------------------------------------------
// Kernel 0: fp32 -> bf16 convert (x), vectorized 4-wide, grid-stride.
// ---------------------------------------------------------------------------
__global__ __launch_bounds__(256)
void f32_to_bf16(const float* __restrict__ in, __bf16* __restrict__ out, long long nvec4) {
  long long i = (long long)blockIdx.x * blockDim.x + threadIdx.x;
  long long stride = (long long)gridDim.x * blockDim.x;
  for (; i < nvec4; i += stride) {
    float4 f = reinterpret_cast<const float4*>(in)[i];
    __bf16* o = out + i * 4;
    o[0] = (__bf16)f.x; o[1] = (__bf16)f.y; o[2] = (__bf16)f.z; o[3] = (__bf16)f.w;
  }
}

// ---------------------------------------------------------------------------
// Tiled bf16 WMMA GEMM: C[M=8192, N=1024] = A[M,1024](bf16) * W[1024,N](f32->bf16)
// 256 threads (8 waves), 256x128 block tile, BK=32. Wave tile: 32x128
// (2 m-tiles x 8 n-tiles = 16 WMMA per k-step per wave).
// A tile staged via async global->LDS b128; W tile converted f32->bf16 and
// transposed into Bs[n][k] through VGPRs.
// OUT_MODE 0: bf16 head-split [b][h][n][hd] (QKV);  OUT_MODE 1: f32 row-major.
// ---------------------------------------------------------------------------
template <int OUT_MODE>
__global__ __launch_bounds__(256)
void gemm_xw(const __bf16* __restrict__ A, const float* __restrict__ W,
             void* __restrict__ outp) {
  __shared__ __bf16 As[256][40];  // [m][k], 80B row pitch (16B aligned)
  __shared__ __bf16 Bs[128][40];  // [n][k] (W transposed during staging)

  const int tid   = threadIdx.x;
  const int lane  = tid & 31;
  const int wave  = tid >> 5;    // 0..7 -> 32-row strip
  const int lrow  = lane & 15;   // M index (A/C) or N index (B)
  const int lhalf = lane >> 4;   // which 16-lane half
  const int mBase = blockIdx.y * 256;
  const int nBase = blockIdx.x * 128;

  v8f c[2][8];
#pragma unroll
  for (int mt = 0; mt < 2; ++mt)
#pragma unroll
    for (int nt = 0; nt < 8; ++nt) c[mt][nt] = (v8f)0.0f;

  const int kk0  = tid >> 5;  // 0..7
  const int vidx = tid & 31;  // float4 index across 128 cols

  for (int k0 = 0; k0 < DMODEL; k0 += 32) {
    // ---- stage A tile: 256 rows x 32 cols bf16, one row per thread, async ----
    {
      const __bf16* src = A + (size_t)(mBase + tid) * DMODEL + k0;
      async_copy64((unsigned)(uintptr_t)&As[tid][0], src);
    }
    // ---- stage W tile transposed: Bs[n][k] = bf16(W[k0+k][nBase+n]) ----
#pragma unroll 4
    for (int kk = kk0; kk < 32; kk += 8) {
      float4 w = *reinterpret_cast<const float4*>(
          W + (size_t)(k0 + kk) * PROJ + nBase + vidx * 4);
      Bs[vidx * 4 + 0][kk] = (__bf16)w.x;
      Bs[vidx * 4 + 1][kk] = (__bf16)w.y;
      Bs[vidx * 4 + 2][kk] = (__bf16)w.z;
      Bs[vidx * 4 + 3][kk] = (__bf16)w.w;
    }
    async_wait0();
    __syncthreads();

    // ---- A fragments: lane half selects K-chunks {koff..koff+7, 16+koff..} ----
    V16 af[2];
#pragma unroll
    for (int mt = 0; mt < 2; ++mt) {
      const __bf16* r = &As[wave * 32 + mt * 16 + lrow][0];
      af[mt].h[0] = *(const v8bf*)(r + lhalf * 8);
      af[mt].h[1] = *(const v8bf*)(r + 16 + lhalf * 8);
    }
    // ---- B fragments: lane half selects K-chunks {koff..koff+7, 16+koff..} ----
#pragma unroll
    for (int nt = 0; nt < 8; ++nt) {
      V16 bfr;
      const __bf16* r = &Bs[nt * 16 + lrow][0];
      bfr.h[0] = *(const v8bf*)(r + lhalf * 8);
      bfr.h[1] = *(const v8bf*)(r + 16 + lhalf * 8);
#pragma unroll
      for (int mt = 0; mt < 2; ++mt)
        c[mt][nt] = wmma_bf16(af[mt].v, bfr.v, c[mt][nt]);
    }
    __syncthreads();
  }

  // ---- store: C layout VGPR r -> row (r + 8*lhalf), col lrow ----
#pragma unroll
  for (int mt = 0; mt < 2; ++mt) {
#pragma unroll
    for (int nt = 0; nt < 8; ++nt) {
#pragma unroll
      for (int r = 0; r < 8; ++r) {
        const int gm = mBase + wave * 32 + mt * 16 + r + 8 * lhalf;
        const int gn = nBase + nt * 16 + lrow;
        const float val = c[mt][nt][r];
        if (OUT_MODE == 0) {
          const int b = gm >> 11, nn = gm & (NSEQ - 1);
          const int h = gn >> 7, hd = gn & (HDIM - 1);
          ((__bf16*)outp)[(((size_t)b * NHEADS + h) * NSEQ + nn) * HDIM + hd] =
              (__bf16)val;
        } else {
          ((float*)outp)[(size_t)gm * PROJ + gn] = val;
        }
      }
    }
  }
}

// ---------------------------------------------------------------------------
// Flash attention. Grid: (NSEQ/128, B*H). 256 threads = 8 waves, 16 q-rows each.
// Streams 64-key blocks; S = Q K^T (16 WMMA), online softmax, O += P V (16 WMMA).
// K tile staged async; V tile transposed through VGPRs.
// ---------------------------------------------------------------------------
__global__ __launch_bounds__(256)
void flash_attn(const __bf16* __restrict__ Q, const __bf16* __restrict__ K,
                const __bf16* __restrict__ V, __bf16* __restrict__ O) {
  __shared__ __bf16 Ks[64][136];    // [key][hd]   (272B pitch, 16B aligned)
  __shared__ __bf16 Vt[128][72];    // [hd][key]   (transposed V, 144B pitch)
  __shared__ __bf16 Ps[8][16][72];  // per-wave P scratch (C-layout -> A-layout)

  const int tid   = threadIdx.x;
  const int lane  = tid & 31;
  const int wave  = tid >> 5;
  const int lrow  = lane & 15;
  const int lhalf = lane >> 4;
  const int bh    = blockIdx.y;          // b*8 + h
  const int qBase = blockIdx.x * 128;

  const __bf16* Kb = K + (size_t)bh * NSEQ * HDIM;
  const __bf16* Vb = V + (size_t)bh * NSEQ * HDIM;

  // ---- Q fragments (held in registers for the whole kernel) ----
  V16 qf[4];
  {
    const __bf16* qrow =
        Q + ((size_t)bh * NSEQ + qBase + wave * 16 + lrow) * HDIM;
#pragma unroll
    for (int ks = 0; ks < 4; ++ks) {
      qf[ks].h[0] = *(const v8bf*)(qrow + ks * 32 + lhalf * 8);
      qf[ks].h[1] = *(const v8bf*)(qrow + ks * 32 + 16 + lhalf * 8);
    }
  }

  v8f o[8];
#pragma unroll
  for (int nt = 0; nt < 8; ++nt) o[nt] = (v8f)0.0f;
  float mrow[8], lrowsum[8];
#pragma unroll
  for (int r = 0; r < 8; ++r) { mrow[r] = -1e30f; lrowsum[r] = 0.0f; }

  const int crow = tid >> 2;        // 0..63 (cooperative load row)
  const int ccol = (tid & 3) * 32;  // 0,32,64,96

  for (int j0 = 0; j0 < NSEQ; j0 += 64) {
    // ---- stage K block row-major (async direct copy: 64B per thread) ----
    async_copy64((unsigned)(uintptr_t)&Ks[crow][ccol],
                 Kb + (size_t)(j0 + crow) * HDIM + ccol);
    // ---- stage V block transposed (through VGPRs) ----
    {
      const __bf16* src = Vb + (size_t)(j0 + crow) * HDIM + ccol;
#pragma unroll
      for (int ch = 0; ch < 4; ++ch) {
        v8bf a = *(const v8bf*)(src + ch * 8);
#pragma unroll
        for (int i = 0; i < 8; ++i) Vt[ccol + ch * 8 + i][crow] = a[i];
      }
    }
    async_wait0();
    __syncthreads();

    // ---- S = Q * K^T : four 16-key subtiles, 4 k-steps over HD=128 ----
    v8f s[4];
#pragma unroll
    for (int kt = 0; kt < 4; ++kt) s[kt] = (v8f)0.0f;
#pragma unroll
    for (int ks = 0; ks < 4; ++ks) {
#pragma unroll
      for (int kt = 0; kt < 4; ++kt) {
        V16 bk;  // B[k=hd][n=key]: lane = key, hd chunks {8h..8h+7, 16+8h..} per half
        const __bf16* r = &Ks[kt * 16 + lrow][0];
        bk.h[0] = *(const v8bf*)(r + ks * 32 + lhalf * 8);
        bk.h[1] = *(const v8bf*)(r + ks * 32 + 16 + lhalf * 8);
        s[kt] = wmma_bf16(qf[ks].v, bk.v, s[kt]);
      }
    }

    // ---- online softmax (rows live across 16-lane halves) ----
    float p[4][8];
#pragma unroll
    for (int r = 0; r < 8; ++r) {
      float v0 = (float)s[0][r] * SM_SCALE;
      float v1 = (float)s[1][r] * SM_SCALE;
      float v2 = (float)s[2][r] * SM_SCALE;
      float v3 = (float)s[3][r] * SM_SCALE;
      const float mloc = half_max16(fmaxf(fmaxf(v0, v1), fmaxf(v2, v3)));
      const float mnew = fmaxf(mrow[r], mloc);
      const float alpha = exp2f((mrow[r] - mnew) * L2E);
      p[0][r] = exp2f((v0 - mnew) * L2E);
      p[1][r] = exp2f((v1 - mnew) * L2E);
      p[2][r] = exp2f((v2 - mnew) * L2E);
      p[3][r] = exp2f((v3 - mnew) * L2E);
      const float ssum =
          half_sum16((p[0][r] + p[1][r]) + (p[2][r] + p[3][r]));
      lrowsum[r] = lrowsum[r] * alpha + ssum;
      mrow[r] = mnew;
#pragma unroll
      for (int nt = 0; nt < 8; ++nt) o[nt][r] = o[nt][r] * alpha;
    }

    // ---- C-layout P -> A-layout via per-wave LDS scratch ----
#pragma unroll
    for (int r = 0; r < 8; ++r) {
      const int mr = r + 8 * lhalf;
#pragma unroll
      for (int kt = 0; kt < 4; ++kt)
        Ps[wave][mr][kt * 16 + lrow] = (__bf16)p[kt][r];
    }
    __builtin_amdgcn_wave_barrier();  // same-wave LDS ordering fence
    V16 pa[2];
#pragma unroll
    for (int kc = 0; kc < 2; ++kc) {
      pa[kc].h[0] = *(const v8bf*)(&Ps[wave][lrow][kc * 32 + lhalf * 8]);
      pa[kc].h[1] = *(const v8bf*)(&Ps[wave][lrow][kc * 32 + 16 + lhalf * 8]);
    }

    // ---- O += P * V (two 32-key k-steps per hd tile) ----
#pragma unroll
    for (int nt = 0; nt < 8; ++nt) {
      const __bf16* r = &Vt[nt * 16 + lrow][0];
#pragma unroll
      for (int kc = 0; kc < 2; ++kc) {
        V16 vb;  // B[k=key][n=hd]: lane = hd, key chunks {8h..8h+7, 16+8h..} per half
        vb.h[0] = *(const v8bf*)(r + kc * 32 + lhalf * 8);
        vb.h[1] = *(const v8bf*)(r + kc * 32 + 16 + lhalf * 8);
        o[nt] = wmma_bf16(pa[kc].v, vb.v, o[nt]);
      }
    }
    __syncthreads();
  }

  // ---- normalize + store to [b][n][h*128+hd] (bf16) ----
  const int b = bh >> 3, h = bh & 7;
#pragma unroll
  for (int r = 0; r < 8; ++r) {
    const float inv = 1.0f / lrowsum[r];
    const int gq = qBase + wave * 16 + r + 8 * lhalf;
#pragma unroll
    for (int nt = 0; nt < 8; ++nt) {
      O[((size_t)b * NSEQ + gq) * PROJ + h * HDIM + nt * 16 + lrow] =
          (__bf16)(o[nt][r] * inv);
    }
  }
}

// ---------------------------------------------------------------------------
// Host-side launcher. Inputs: x, Wq, Wk, Wv, Wo (all fp32). Output: fp32.
// Workspace layout (16 MB regions): xb | Q | K | V | AO  (80 MB total).
// ---------------------------------------------------------------------------
extern "C" void kernel_launch(void* const* d_in, const int* in_sizes, int n_in,
                              void* d_out, int out_size, void* d_ws, size_t ws_size,
                              hipStream_t stream) {
  (void)in_sizes; (void)n_in; (void)out_size; (void)ws_size;
  const float* x  = (const float*)d_in[0];
  const float* Wq = (const float*)d_in[1];
  const float* Wk = (const float*)d_in[2];
  const float* Wv = (const float*)d_in[3];
  const float* Wo = (const float*)d_in[4];

  char* ws = (char*)d_ws;
  const size_t REG = (size_t)BATCH * NSEQ * DMODEL * sizeof(__bf16);  // 16 MB
  __bf16* xb = (__bf16*)(ws);
  __bf16* Qb = (__bf16*)(ws + 1 * REG);
  __bf16* Kb = (__bf16*)(ws + 2 * REG);
  __bf16* Vb = (__bf16*)(ws + 3 * REG);
  __bf16* AO = (__bf16*)(ws + 4 * REG);

  const long long nvec4 = (long long)BATCH * NSEQ * DMODEL / 4;
  f32_to_bf16<<<dim3(2048), dim3(256), 0, stream>>>(x, xb, nvec4);

  const dim3 gemmGrid(PROJ / 128, (BATCH * NSEQ) / 256);  // (8, 32)
  gemm_xw<0><<<gemmGrid, dim3(256), 0, stream>>>(xb, Wq, Qb);
  gemm_xw<0><<<gemmGrid, dim3(256), 0, stream>>>(xb, Wk, Kb);
  gemm_xw<0><<<gemmGrid, dim3(256), 0, stream>>>(xb, Wv, Vb);

  flash_attn<<<dim3(NSEQ / 128, BATCH * NHEADS), dim3(256), 0, stream>>>(
      Qb, Kb, Vb, AO);

  gemm_xw<1><<<gemmGrid, dim3(256), 0, stream>>>(AO, Wo, d_out);
}
